// KAN_GLUExpert_86517821216092
// MI455X (gfx1250) — matrix operand order repaired and checked by reading, not compiled, not run on hardware
//
#include <hip/hip_runtime.h>

// ---------------------------------------------------------------------------
// KAN GLU expert for MI455X (gfx1250, wave32, WMMA + async global->LDS).
//   kan_linear == one GEMM over augmented K = 9*in  (silu + 8 spline bases)
//   layers 1&2 share activations -> fused dual-accumulator GLU GEMM
//   bf16 WMMA (v_wmma_f32_16x16x32_bf16), fp32 accumulate
//   A tiles staged with global_load_async_to_lds_b128, double-buffered LDS
//   All 4 A sub-fragments preloaded per k-step -> single dscnt wait, 8 WMMAs
// ---------------------------------------------------------------------------

typedef __attribute__((ext_vector_type(16))) __bf16 v16bf;
typedef __attribute__((ext_vector_type(8)))  float  v8f;

#define SPLINE_K 3
#define GRIDSZ   5
#define NB       8
#define NAUG     9
#define D_MODEL  1024
#define D_FF     4096
#define NTOK     4096
#define TCHUNK   1024

__device__ __forceinline__ float knot_at(int t) {
  return (float)(t - SPLINE_K) * 0.4f - 1.0f;
}

__device__ __forceinline__ void kan_features(float x, float* f) {
  f[0] = x / (1.0f + __expf(-x));
  float bas[GRIDSZ + 2 * SPLINE_K];
#pragma unroll
  for (int t = 0; t < GRIDSZ + 2 * SPLINE_K; ++t)
    bas[t] = (x >= knot_at(t) && x < knot_at(t + 1)) ? 1.0f : 0.0f;
#pragma unroll
  for (int j = 1; j <= SPLINE_K; ++j) {
    const float inv = 1.0f / (0.4f * (float)j);
#pragma unroll
    for (int t = 0; t < GRIDSZ + 2 * SPLINE_K - j; ++t) {
      float left  = (x - knot_at(t)) * inv;
      float right = (knot_at(t + j + 1) - x) * inv;
      bas[t] = left * bas[t] + right * bas[t + 1];
    }
  }
#pragma unroll
  for (int c = 0; c < NB; ++c) f[1 + c] = bas[c];
}

__global__ __launch_bounds__(256) void expand_kan_act(
    const float* __restrict__ X, __bf16* __restrict__ A, int nElems) {
  int idx = blockIdx.x * 256 + threadIdx.x;
  if (idx >= nElems) return;
  float f[NAUG];
  kan_features(X[idx], f);
  __bf16* dst = A + (size_t)idx * NAUG;
#pragma unroll
  for (int c = 0; c < NAUG; ++c) dst[c] = (__bf16)f[c];
}

__global__ __launch_bounds__(256) void convert_kan_w(
    const float* __restrict__ baseW, const float* __restrict__ splineW,
    __bf16* __restrict__ Waug, int nElems) {
  int idx = blockIdx.x * 256 + threadIdx.x;
  if (idx >= nElems) return;
  __bf16* dst = Waug + (size_t)idx * NAUG;
  dst[0] = (__bf16)baseW[idx];
  const float* s = splineW + (size_t)idx * NB;
#pragma unroll
  for (int c = 0; c < NB; ++c) dst[1 + c] = (__bf16)s[c];
}

// one 16B async global->LDS copy (CDNA5, ASYNCcnt-tracked)
__device__ __forceinline__ void async_copy_b128(unsigned ldsOff, const void* g) {
  asm volatile("global_load_async_to_lds_b128 %0, %1, off"
               :: "v"(ldsOff), "v"(g) : "memory");
}
__device__ __forceinline__ void wait_async0() {
  asm volatile("s_wait_asynccnt 0x0" ::: "memory");
}

// ---------------------------------------------------------------------------
// WMMA GEMM:  Out(M,N) [+GLU elementwise product] = A(M,K) x W(N,K)^T.
// Block: 256 thr = 8 waves; block tile 64(M) x 128(N); wave tile 64x16.
// ---------------------------------------------------------------------------
template <bool GLU>
__global__ __launch_bounds__(256) void kan_wmma_gemm(
    const __bf16* __restrict__ A,  // (M, K)
    const __bf16* __restrict__ W1, // (N, K)
    const __bf16* __restrict__ W2, // (N, K), GLU only
    float* __restrict__ Out,       // (M, N)
    int M, int N, int K) {
  __shared__ __align__(16) unsigned int sA[2][64][16]; // 2 x (64 rows x 32 bf16)

  const int tid  = threadIdx.x;
  const int wave = tid >> 5;
  const int lane = tid & 31;
  const int lr   = lane & 15;
  const int lh   = lane >> 4;
  const int m0   = blockIdx.y << 6;
  const int ncol = (blockIdx.x << 7) + (wave << 4) + lr;

  // staging: thread t copies one 16B chunk; row = t>>2, quarter = t&3
  const int srow = tid >> 2;
  const int sq   = tid & 3;
  const __bf16* gsrcBase = A + (size_t)(m0 + srow) * K + (sq << 3);
  const unsigned lds0 = (unsigned)(size_t)&sA[0][srow][sq << 2];
  const unsigned lds1 = (unsigned)(size_t)&sA[1][srow][sq << 2];

  const __bf16* wrow1 = W1 + (size_t)ncol * K + (lh << 4);
  const __bf16* wrow2 = W2 + (size_t)ncol * K + (lh << 4);

  v8f acc1[4] = {v8f{}, v8f{}, v8f{}, v8f{}};
  v8f acc2[4] = {v8f{}, v8f{}, v8f{}, v8f{}};

  async_copy_b128(lds0, (const void*)gsrcBase); // prologue: stage k0 = 0

  for (int k0 = 0; k0 < K; k0 += 32) {
    const int p = (k0 >> 5) & 1;
    wait_async0();        // my buffer-p staging writes are done
    __syncthreads();      // everyone's done; prev-iter reads retired

    if (k0 + 32 < K)      // overlap next A tile copy with this compute
      async_copy_b128(p ? lds0 : lds1, (const void*)(gsrcBase + k0 + 32));

    // B fragments: lanes 0-15 K=k0..k0+15, lanes 16-31 K=k0+16..k0+31
    const uint4* bp1 = reinterpret_cast<const uint4*>(wrow1 + k0);
    union { uint4 u[2]; v16bf v; } bf1;
    bf1.u[0] = bp1[0]; bf1.u[1] = bp1[1];
    union { uint4 u[2]; v16bf v; } bf2;
    if (GLU) {
      const uint4* bp2 = reinterpret_cast<const uint4*>(wrow2 + k0);
      bf2.u[0] = bp2[0]; bf2.u[1] = bp2[1];
    }
    __builtin_prefetch((const void*)(wrow1 + k0 + 128), 0, 1);

    // preload ALL four A sub-fragments -> one dscnt wait, then 8 WMMAs
    const unsigned int(*tile)[16] = sA[p];
    union { uint4 u[2]; v16bf v; } af[4];
#pragma unroll
    for (int s = 0; s < 4; ++s) {
      af[s].u[0] = *reinterpret_cast<const uint4*>(&tile[(s << 4) + lr][lh << 3]);
      af[s].u[1] = *reinterpret_cast<const uint4*>(&tile[(s << 4) + lr][(lh << 3) + 4]);
    }
#pragma unroll
    for (int s = 0; s < 4; ++s) {
      acc1[s] = __builtin_amdgcn_wmma_f32_16x16x32_bf16(
          false, af[s].v, false, bf1.v, (short)0, acc1[s], false, false);
      if (GLU)
        acc2[s] = __builtin_amdgcn_wmma_f32_16x16x32_bf16(
            false, af[s].v, false, bf2.v, (short)0, acc2[s], false, false);
    }
  }

  // D layout per ISA: vgpr v -> row v (lanes 0-15) / v+8 (lanes 16-31); col = lr
#pragma unroll
  for (int s = 0; s < 4; ++s) {
#pragma unroll
    for (int v = 0; v < 8; ++v) {
      float h = GLU ? acc1[s][v] * acc2[s][v] : acc1[s][v];
      Out[(size_t)(m0 + (s << 4) + v + (lh << 3)) * N + ncol] = h;
    }
  }
}

// ---------------------------------------------------------------------------
extern "C" void kernel_launch(void* const* d_in, const int* in_sizes, int n_in,
                              void* d_out, int out_size, void* d_ws, size_t ws_size,
                              hipStream_t stream) {
  (void)in_sizes; (void)n_in; (void)out_size; (void)ws_size;
  const float* x   = (const float*)d_in[0];
  const float* bw1 = (const float*)d_in[1];
  const float* sw1 = (const float*)d_in[2];
  const float* bw2 = (const float*)d_in[3];
  const float* sw2 = (const float*)d_in[4];
  const float* bw3 = (const float*)d_in[5];
  const float* sw3 = (const float*)d_in[6];
  // grids (d_in[7], d_in[8]) are uniform by construction; knots recomputed on device

  const size_t SEG = (size_t)D_FF * D_MODEL * NAUG * sizeof(__bf16); // 75,497,472
  char*   ws  = (char*)d_ws;
  __bf16* A1  = (__bf16*)(ws);           // later: A3 token-chunk (same size)
  __bf16* W1a = (__bf16*)(ws + SEG);     // later: W3aug (same size)
  __bf16* W2a = (__bf16*)(ws + 2 * SEG);
  float*  H   = (float*)(ws + 3 * SEG);  // (4096, 4096) f32; peak ws ~294 MB

  expand_kan_act<<<(NTOK * D_MODEL) / 256, 256, 0, stream>>>(x, A1, NTOK * D_MODEL);
  convert_kan_w<<<(D_FF * D_MODEL) / 256, 256, 0, stream>>>(bw1, sw1, W1a, D_FF * D_MODEL);
  convert_kan_w<<<(D_FF * D_MODEL) / 256, 256, 0, stream>>>(bw2, sw2, W2a, D_FF * D_MODEL);

  dim3 g1(D_FF / 128, NTOK / 64);
  kan_wmma_gemm<true><<<g1, 256, 0, stream>>>(A1, W1a, W2a, H,
                                              NTOK, D_FF, D_MODEL * NAUG);

  convert_kan_w<<<(D_MODEL * D_FF) / 256, 256, 0, stream>>>(bw3, sw3, W1a, D_MODEL * D_FF);

  for (int c = 0; c < NTOK / TCHUNK; ++c) {
    expand_kan_act<<<(TCHUNK * D_FF) / 256, 256, 0, stream>>>(
        H + (size_t)c * TCHUNK * D_FF, A1, TCHUNK * D_FF);
    dim3 g3(D_MODEL / 128, TCHUNK / 64);
    kan_wmma_gemm<false><<<g3, 256, 0, stream>>>(
        A1, W1a, W1a, (float*)d_out + (size_t)c * TCHUNK * D_MODEL,
        TCHUNK, D_MODEL, D_FF * NAUG);
  }
}
